// SS2D_41266045780052
// MI455X (gfx1250) — matrix operand order repaired and checked by reading, not compiled, not run on hardware
//
#include <hip/hip_runtime.h>
#include <hip/hip_bf16.h>
#include <math.h>

// ---------------- problem constants ----------------
#define Bsz   2
#define Dm    96
#define Ei    192           // D_INNER
#define Nst   16            // D_STATE
#define Rr    6             // DT_RANK
#define Hh    64
#define Ww    64
#define Ll    4096          // H*W
#define Mtot  (Bsz*Ll)      // 8192 rows for (b,l)-flattened GEMMs
#define NDIR  4
#define KB    (NDIR*Bsz)    // 8 scan sequences
#define PROJ  (Rr + 2*Nst)  // 38
#define NSEG  32
#define SEGL  (Ll/NSEG)     // 128
#define CHUNK 8

typedef __attribute__((ext_vector_type(16))) _Float16 v16h;
typedef __attribute__((ext_vector_type(8)))  float    v8f;

#ifndef __has_builtin
#define __has_builtin(x) 0
#endif
#if __has_builtin(__builtin_amdgcn_global_load_async_to_lds_b32)
#define HAVE_ASYNC_CP 1
#else
#define HAVE_ASYNC_CP 0
#endif
#if __has_builtin(__builtin_amdgcn_s_wait_asynccnt)
#define WAIT_ASYNC() __builtin_amdgcn_s_wait_asynccnt(0)
#else
#define WAIT_ASYNC() asm volatile("s_wait_asynccnt 0" ::: "memory")
#endif

// ---------------- WMMA fragment builders (CDNA5 wave32 layouts) ----------------
// A 16x32 f16 (MxK), row-major in LDS: lanes 0-15 -> M=lane, K = 0..15;
// lanes 16-31 -> same M, K = 16..31.  Per-lane reads are 16 contiguous halves.
__device__ inline v16h load_frag_a(const _Float16* As /*16 x 32, row-major*/, int lane) {
  int row = lane & 15, grp = lane >> 4;
  v16h a;
#pragma unroll
  for (int j = 0; j < 16; ++j) {
    int k = grp * 16 + j;
    a[j] = As[row * 32 + k];
  }
  return a;
}
// B 32x16 f16 (KxN) stored N-MAJOR in LDS as Bn[n][k] (16 x 32): lane's fragment
// (N=lane&15, K = grp*16 + 0..15) is 16 CONTIGUOUS halves -> 2x ds_load_b128.
__device__ inline v16h load_frag_bn(const _Float16* Bn /*16 x 32, n-major*/, int lane) {
  int col = lane & 15, grp = lane >> 4;
  v16h b;
#pragma unroll
  for (int j = 0; j < 16; ++j) b[j] = Bn[col * 32 + grp * 16 + j];
  return b;
}

__device__ inline float silu_f(float x) { return x / (1.f + __expf(-x)); }
__device__ inline float softplus_f(float x) { return (x > 20.f) ? x : log1pf(__expf(x)); }

// scan-order step l -> row-major spatial index, per direction
__device__ inline int scan_pos(int dir, int l) {
  int ll = (dir & 1) ? (Ll - 1 - l) : l;
  if (dir >= 2) { int h = ll & (Hh - 1); int w = ll >> 6; return h * Ww + w; }
  return ll;
}

// ---------------- small utility kernels ----------------
__global__ void f32_to_f16_kernel(const float* __restrict__ in, _Float16* __restrict__ out, int n) {
  int i = blockIdx.x * 256 + threadIdx.x;
  if (i < n) out[i] = (_Float16)in[i];
}

__global__ void zero_f32_kernel(float* __restrict__ p, int n) {
  int i = blockIdx.x * 256 + threadIdx.x;
  if (i < n) p[i] = 0.f;
}

// ---------------- 1) in_proj GEMM: xz = x_flat @ W^T  (M=8192, K=96, N=384) ----------------
// Each wave computes a 16x32 output strip (2 WMMAs / K-step, A fragment reused).
// cols [0,192)  -> xb_pre (B,E,L) for the conv; cols [192,384) -> z (B,L,E) for the gate.
__global__ __launch_bounds__(32) void gemm_inproj(const float* __restrict__ x,
                                                  const _Float16* __restrict__ w16,
                                                  float* __restrict__ xb_pre,
                                                  float* __restrict__ z) {
  __shared__ _Float16 As[16 * 32];
  __shared__ _Float16 Bs2[32 * 32];     // two 16-col tiles, n-major
  int lane = threadIdx.x;
  int m0 = blockIdx.x * 16, n0 = blockIdx.y * 32;
  v8f c0 = {}, c1 = {};
  for (int k0 = 0; k0 < Dm; k0 += 32) {       // 96 = 3 * 32, no tail
    for (int t = lane; t < 16 * 32; t += 32) {
      int r = t >> 5, k = t & 31;
      int m = m0 + r, b = m >> 12, l = m & (Ll - 1);
      As[t] = (_Float16)x[((size_t)b * Dm + (k0 + k)) * Ll + l];
    }
    for (int t = lane; t < 32 * 32; t += 32) {
      int n = t >> 5, k = t & 31;             // n relative 0..31
      Bs2[t] = w16[(size_t)(n0 + n) * Dm + (k0 + k)];   // W is (384,96) row-major
    }
    __syncthreads();
    v16h a  = load_frag_a(As, lane);
    v16h b0 = load_frag_bn(Bs2, lane);
    v16h b1 = load_frag_bn(Bs2 + 16 * 32, lane);
    c0 = __builtin_amdgcn_wmma_f32_16x16x32_f16(false, a, false, b0, (short)0, c0, false, false);
    c1 = __builtin_amdgcn_wmma_f32_16x16x32_f16(false, a, false, b1, (short)0, c1, false, false);
    __syncthreads();
  }
  int grp = lane >> 4, col = lane & 15;
#pragma unroll
  for (int r = 0; r < 8; ++r) {
    int m = m0 + r + grp * 8;
    int b = m >> 12, l = m & (Ll - 1);
#pragma unroll
    for (int t = 0; t < 2; ++t) {
      int n = n0 + t * 16 + col;
      float v = t ? c1[r] : c0[r];
      if (n < Ei) xb_pre[((size_t)b * Ei + n) * Ll + l] = v;
      else        z[((size_t)b * Ll + l) * Ei + (n - Ei)] = v;
    }
  }
}

// ---------------- 2) depthwise 3x3 conv + bias + SiLU ----------------
__global__ void conv_silu(const float* __restrict__ xb_pre, const float* __restrict__ cw,
                          const float* __restrict__ cb, float* __restrict__ xbc) {
  int idx = blockIdx.x * 256 + threadIdx.x;
  if (idx >= Bsz * Ei * Ll) return;
  int l = idx & (Ll - 1);
  int be = idx >> 12;           // b*Ei + e
  int e = be % Ei;
  int h = l >> 6, w = l & 63;
  float acc = cb[e];
#pragma unroll
  for (int dy = -1; dy <= 1; ++dy) {
    int hy = h + dy;
    if (hy < 0 || hy >= Hh) continue;
#pragma unroll
    for (int dx = -1; dx <= 1; ++dx) {
      int wx = w + dx;
      if (wx < 0 || wx >= Ww) continue;
      acc += xb_pre[((size_t)be << 12) + hy * Ww + wx] * cw[e * 9 + (dy + 1) * 3 + (dx + 1)];
    }
  }
  xbc[idx] = silu_f(acc);
}

// ---------------- 3) x_proj GEMM per direction (M=8192, K=192, N=38 pad 64) ----------------
// A[l][e] gathered from xbc with direction-dependent indexing (xs never materialized).
__global__ __launch_bounds__(32) void gemm_xproj(const float* __restrict__ xbc,
                                                 const _Float16* __restrict__ xpw16,
                                                 float* __restrict__ xdbl) {
  __shared__ _Float16 As[16 * 32];
  __shared__ _Float16 Bs2[32 * 32];
  int lane = threadIdx.x;
  int dir = blockIdx.x >> 9;
  int mtile = blockIdx.x & 511;
  int m0 = mtile * 16, n0 = blockIdx.y * 32;
  v8f c0 = {}, c1 = {};
  for (int k0 = 0; k0 < Ei; k0 += 32) {       // 192 = 6 * 32
    for (int t = lane; t < 16 * 32; t += 32) {
      int r = t >> 5, k = t & 31;
      int m = m0 + r, b = m >> 12, l = m & (Ll - 1);
      int p = scan_pos(dir, l);
      As[t] = (_Float16)xbc[((size_t)b * Ei + (k0 + k)) * Ll + p];
    }
    for (int t = lane; t < 32 * 32; t += 32) {
      int n = t >> 5, k = t & 31;
      int ng = n0 + n;
      Bs2[t] = (ng < PROJ) ? xpw16[((size_t)dir * Ei + (k0 + k)) * PROJ + ng] : (_Float16)0.f;
    }
    __syncthreads();
    v16h a  = load_frag_a(As, lane);
    v16h b0 = load_frag_bn(Bs2, lane);
    v16h b1 = load_frag_bn(Bs2 + 16 * 32, lane);
    c0 = __builtin_amdgcn_wmma_f32_16x16x32_f16(false, a, false, b0, (short)0, c0, false, false);
    c1 = __builtin_amdgcn_wmma_f32_16x16x32_f16(false, a, false, b1, (short)0, c1, false, false);
    __syncthreads();
  }
  int grp = lane >> 4, col = lane & 15;
#pragma unroll
  for (int r = 0; r < 8; ++r) {
    int m = m0 + r + grp * 8;
    int b = m >> 12, l = m & (Ll - 1);
#pragma unroll
    for (int t = 0; t < 2; ++t) {
      int n = n0 + t * 16 + col;
      if (n < PROJ)
        xdbl[(((size_t)dir * Bsz + b) * Ll + l) * PROJ + n] = t ? c1[r] : c0[r];
    }
  }
}

// ---------------- 4) segmented selective scan ----------------
// Pass A (FINAL=false): per segment with h0=0 compute (prod deltaA, h_end) per (kb,e,n).
// Pass B: serial carry over 32 segments per (kb,e,n).
// Pass C (FINAL=true): replay with correct h0, emit y (+= into yo at spatial pos).
template <bool FINAL>
__global__ __launch_bounds__(Ei) void scan_seg(const float* __restrict__ xdbl,
                                               const float* __restrict__ xbc,
                                               const float* __restrict__ A_log,
                                               const float* __restrict__ dtw,
                                               const float* __restrict__ dtb,
                                               const float* __restrict__ Dsk,
                                               float* __restrict__ aprod,
                                               float* __restrict__ hend,
                                               const float* __restrict__ hstart,
                                               float* __restrict__ yo) {
  int kb = blockIdx.x >> 5;
  int seg = blockIdx.x & 31;
  int dir = kb / Bsz, b = kb % Bsz;
  int e = threadIdx.x;

  float Arow[Nst];
#pragma unroll
  for (int n = 0; n < Nst; ++n) Arow[n] = -__expf(A_log[e * Nst + n]);
  float wdt[Rr];
#pragma unroll
  for (int r = 0; r < Rr; ++r) wdt[r] = dtw[((size_t)dir * Ei + e) * Rr + r];
  float bias = dtb[dir * Ei + e];
  float dsk = Dsk[e];

  float h[Nst], ap[Nst];
  size_t sidx = (((size_t)(kb * NSEG + seg)) * Ei + e) * Nst;
#pragma unroll
  for (int n = 0; n < Nst; ++n) {
    h[n] = FINAL ? hstart[sidx + n] : 0.f;
    ap[n] = 1.f;
  }

  __shared__ float sm[CHUNK * PROJ];
  const float* base = xdbl + (size_t)kb * Ll * PROJ;
  int l0beg = seg * SEGL;

  for (int l0 = l0beg; l0 < l0beg + SEGL; l0 += CHUNK) {
    __syncthreads();
#if HAVE_ASYNC_CP
    // CDNA5 async DMA: per-lane global -> LDS copy, tracked with ASYNCcnt.
    for (int t = threadIdx.x; t < CHUNK * PROJ; t += Ei)
      __builtin_amdgcn_global_load_async_to_lds_b32(
          (__attribute__((address_space(1))) int*)(base + (size_t)l0 * PROJ + t),
          (__attribute__((address_space(3))) int*)(sm + t), 0, 0);
    WAIT_ASYNC();
#else
    for (int t = threadIdx.x; t < CHUNK * PROJ; t += Ei)
      sm[t] = base[(size_t)l0 * PROJ + t];
#endif
    __syncthreads();
    if (l0 + CHUNK < l0beg + SEGL)
      __builtin_prefetch(base + (size_t)(l0 + CHUNK) * PROJ, 0, 0);   // global_prefetch_b8

    for (int j = 0; j < CHUNK; ++j) {
      int l = l0 + j;
      const float* row = sm + j * PROJ;
      float s = bias;
#pragma unroll
      for (int r = 0; r < Rr; ++r) s += row[r] * wdt[r];
      float dt = softplus_f(s);
      int p = scan_pos(dir, l);
      float u = xbc[((size_t)b * Ei + e) * Ll + p];
      float dtu = dt * u;
      float acc = u * dsk;
#pragma unroll
      for (int n = 0; n < Nst; ++n) {
        float dA = __expf(dt * Arow[n]);
        h[n] = dA * h[n] + dtu * row[Rr + n];
        if (!FINAL) ap[n] *= dA;
        if (FINAL) acc += h[n] * row[Rr + Nst + n];
      }
      if (FINAL) atomicAdd(&yo[((size_t)b * Ei + e) * Ll + p], acc);
    }
  }
  if (!FINAL) {
#pragma unroll
    for (int n = 0; n < Nst; ++n) {
      aprod[sidx + n] = ap[n];
      hend[sidx + n] = h[n];
    }
  }
}

__global__ void scan_carry(const float* __restrict__ aprod, const float* __restrict__ hend,
                           float* __restrict__ hstart) {
  int t = blockIdx.x * 256 + threadIdx.x;     // over KB*Ei*Nst = 24576
  if (t >= KB * Ei * Nst) return;
  int kb = t / (Ei * Nst);
  int en = t % (Ei * Nst);
  float c = 0.f;
  for (int s = 0; s < NSEG; ++s) {
    size_t idx = ((size_t)(kb * NSEG + s)) * (Ei * Nst) + en;
    hstart[idx] = c;
    c = aprod[idx] * c + hend[idx];
  }
}

// ---------------- 5) LayerNorm over E + SiLU(z) gate -> f16 activations ----------------
__global__ __launch_bounds__(Ei) void ln_gate(const float* __restrict__ yo,
                                              const float* __restrict__ z,
                                              const float* __restrict__ gamma,
                                              const float* __restrict__ beta,
                                              _Float16* __restrict__ yg) {
  int bl = blockIdx.x;                 // b*Ll + l
  int b = bl >> 12, l = bl & (Ll - 1);
  int e = threadIdx.x;
  float v = yo[((size_t)b * Ei + e) * Ll + l];

  __shared__ float red[Ei];
  __shared__ float mean_s, var_s;
  red[e] = v; __syncthreads();
  if (e < 96) red[e] += red[e + 96]; __syncthreads();
  if (e < 48) red[e] += red[e + 48]; __syncthreads();
  if (e < 24) red[e] += red[e + 24]; __syncthreads();
  if (e < 12) red[e] += red[e + 12]; __syncthreads();
  if (e < 6)  red[e] += red[e + 6];  __syncthreads();
  if (e == 0) mean_s = (red[0] + red[1] + red[2] + red[3] + red[4] + red[5]) * (1.f / Ei);
  __syncthreads();
  float d = v - mean_s;
  red[e] = d * d; __syncthreads();
  if (e < 96) red[e] += red[e + 96]; __syncthreads();
  if (e < 48) red[e] += red[e + 48]; __syncthreads();
  if (e < 24) red[e] += red[e + 24]; __syncthreads();
  if (e < 12) red[e] += red[e + 12]; __syncthreads();
  if (e < 6)  red[e] += red[e + 6];  __syncthreads();
  if (e == 0) var_s = (red[0] + red[1] + red[2] + red[3] + red[4] + red[5]) * (1.f / Ei);
  __syncthreads();

  float yn = d * rsqrtf(var_s + 1e-5f) * gamma[e] + beta[e];
  float zz = z[(size_t)bl * Ei + e];
  yg[(size_t)bl * Ei + e] = (_Float16)(yn * silu_f(zz));
}

// ---------------- 6) out_proj GEMM (M=8192, K=192, N=96) -> d_out (B,D,H,W) ----------------
__global__ __launch_bounds__(32) void gemm_outproj(const _Float16* __restrict__ yg,
                                                   const _Float16* __restrict__ ow16,
                                                   float* __restrict__ out) {
  __shared__ _Float16 As[16 * 32];
  __shared__ _Float16 Bs2[32 * 32];
  int lane = threadIdx.x;
  int m0 = blockIdx.x * 16, n0 = blockIdx.y * 32;
  v8f c0 = {}, c1 = {};
  for (int k0 = 0; k0 < Ei; k0 += 32) {
    for (int t = lane; t < 16 * 32; t += 32) {
      int r = t >> 5, k = t & 31;
      As[t] = yg[(size_t)(m0 + r) * Ei + (k0 + k)];
    }
    for (int t = lane; t < 32 * 32; t += 32) {
      int n = t >> 5, k = t & 31;
      Bs2[t] = ow16[(size_t)(n0 + n) * Ei + (k0 + k)];   // out_proj_w is (96,192)
    }
    __syncthreads();
    v16h a  = load_frag_a(As, lane);
    v16h b0 = load_frag_bn(Bs2, lane);
    v16h b1 = load_frag_bn(Bs2 + 16 * 32, lane);
    c0 = __builtin_amdgcn_wmma_f32_16x16x32_f16(false, a, false, b0, (short)0, c0, false, false);
    c1 = __builtin_amdgcn_wmma_f32_16x16x32_f16(false, a, false, b1, (short)0, c1, false, false);
    __syncthreads();
  }
  int grp = lane >> 4, col = lane & 15;
#pragma unroll
  for (int r = 0; r < 8; ++r) {
    int m = m0 + r + grp * 8;
    int b = m >> 12, l = m & (Ll - 1);
    out[((size_t)b * Dm + (n0 + col)) * Ll + l] = c0[r];
    out[((size_t)b * Dm + (n0 + 16 + col)) * Ll + l] = c1[r];
  }
}

// ---------------- launch ----------------
extern "C" void kernel_launch(void* const* d_in, const int* in_sizes, int n_in,
                              void* d_out, int out_size, void* d_ws, size_t ws_size,
                              hipStream_t stream) {
  const float* x      = (const float*)d_in[0];
  const float* ipw    = (const float*)d_in[1];   // (384,96)
  const float* cw     = (const float*)d_in[2];   // (192,1,3,3)
  const float* cb     = (const float*)d_in[3];   // (192)
  const float* xpw    = (const float*)d_in[4];   // (4,192,38)
  const float* dtw    = (const float*)d_in[5];   // (4,192,6)
  const float* dtb    = (const float*)d_in[6];   // (4,192)
  const float* A_log  = (const float*)d_in[7];   // (192,16)
  const float* Dsk    = (const float*)d_in[8];   // (192)
  const float* lng    = (const float*)d_in[9];   // (192)
  const float* lnb    = (const float*)d_in[10];  // (192)
  const float* opw    = (const float*)d_in[11];  // (96,192)
  float* out = (float*)d_out;

  // workspace layout (all 256B-aligned)
  char* ws = (char*)d_ws;
  size_t off = 0;
  auto alloc = [&](size_t bytes) { size_t o = off; off += (bytes + 255) & ~(size_t)255; return o; };
  const int N_IPW = 2 * Ei * Dm;        // 36864
  const int N_XPW = NDIR * Ei * PROJ;   // 29184
  const int N_OPW = Dm * Ei;            // 18432
  _Float16* w16_in  = (_Float16*)(ws + alloc(N_IPW * 2));
  _Float16* w16_xp  = (_Float16*)(ws + alloc(N_XPW * 2));
  _Float16* w16_out = (_Float16*)(ws + alloc(N_OPW * 2));
  float* xb_pre = (float*)(ws + alloc((size_t)Bsz * Ei * Ll * 4));
  float* zbuf   = (float*)(ws + alloc((size_t)Bsz * Ll * Ei * 4));
  float* xbc    = (float*)(ws + alloc((size_t)Bsz * Ei * Ll * 4));
  float* xdbl   = (float*)(ws + alloc((size_t)KB * Ll * PROJ * 4));
  float* yo     = (float*)(ws + alloc((size_t)Bsz * Ei * Ll * 4));
  _Float16* yg  = (_Float16*)(ws + alloc((size_t)Bsz * Ll * Ei * 2));
  const size_t NSTATE = (size_t)KB * NSEG * Ei * Nst;   // 786432
  float* aprod  = (float*)(ws + alloc(NSTATE * 4));
  float* hend   = (float*)(ws + alloc(NSTATE * 4));
  float* hstart = (float*)(ws + alloc(NSTATE * 4));
  (void)ws_size; (void)in_sizes; (void)n_in; (void)out_size;

  // weight conversions to f16
  f32_to_f16_kernel<<<(N_IPW + 255) / 256, 256, 0, stream>>>(ipw, w16_in, N_IPW);
  f32_to_f16_kernel<<<(N_XPW + 255) / 256, 256, 0, stream>>>(xpw, w16_xp, N_XPW);
  f32_to_f16_kernel<<<(N_OPW + 255) / 256, 256, 0, stream>>>(opw, w16_out, N_OPW);

  // 1) in_proj (each block: 16x32 output strip)
  gemm_inproj<<<dim3(Mtot / 16, (2 * Ei) / 32), 32, 0, stream>>>(x, w16_in, xb_pre, zbuf);
  // 2) conv + SiLU
  conv_silu<<<(Bsz * Ei * Ll + 255) / 256, 256, 0, stream>>>(xb_pre, cw, cb, xbc);
  // 3) x_proj (4 directions fused via gather); N padded 38 -> 64 (2 col-blocks)
  gemm_xproj<<<dim3(NDIR * (Mtot / 16), 2), 32, 0, stream>>>(xbc, w16_xp, xdbl);
  // 4) segmented scan
  zero_f32_kernel<<<(Bsz * Ei * Ll + 255) / 256, 256, 0, stream>>>(yo, Bsz * Ei * Ll);
  scan_seg<false><<<KB * NSEG, Ei, 0, stream>>>(xdbl, xbc, A_log, dtw, dtb, Dsk,
                                                aprod, hend, nullptr, nullptr);
  scan_carry<<<(KB * Ei * Nst + 255) / 256, 256, 0, stream>>>(aprod, hend, hstart);
  scan_seg<true><<<KB * NSEG, Ei, 0, stream>>>(xdbl, xbc, A_log, dtw, dtb, Dsk,
                                               nullptr, nullptr, hstart, yo);
  // 5) LN + gate
  ln_gate<<<Bsz * Ll, Ei, 0, stream>>>(yo, zbuf, lng, lnb, yg);
  // 6) out_proj -> output (B, D, H, W)
  gemm_outproj<<<dim3(Mtot / 16, Dm / 32), 32, 0, stream>>>(yg, w16_out, out);
}
